// EmoNet_12644383719507
// MI455X (gfx1250) — matrix operand order repaired and checked by reading, not compiled, not run on hardware
//
#include <hip/hip_runtime.h>
#include <stdint.h>

// ---------------------------------------------------------------------------
// Types for CDNA5 WMMA / TDM (gfx1250, wave32)
// ---------------------------------------------------------------------------
typedef __attribute__((ext_vector_type(16))) __bf16 v16bf;
typedef __attribute__((ext_vector_type(8)))  __bf16 v8bf;
typedef __attribute__((ext_vector_type(8)))  float  v8f;
typedef __attribute__((ext_vector_type(4)))  unsigned int u32x4;
typedef __attribute__((ext_vector_type(8)))  int i32x8;
typedef __attribute__((ext_vector_type(4)))  int i32x4;

__device__ __forceinline__ __bf16 f2bf(float f) {
    union { float f; unsigned u; } v; v.f = f;
    unsigned r = v.u + 0x7FFFu + ((v.u >> 16) & 1u);   // round-to-nearest-even
    unsigned short h = (unsigned short)(r >> 16);
    return __builtin_bit_cast(__bf16, h);
}

__device__ __forceinline__ v8f wmma_bf16(v16bf a, v16bf b, v8f c) {
    // (neg_a, A, neg_b, B, c_mod, C, reuse_a, reuse_b)
    return __builtin_amdgcn_wmma_f32_16x16x32_bf16(false, a, false, b, (short)0, c, false, false);
}

// A fragment (16x32 bf16) from fp32 row-major activations (conversion in regs).
__device__ __forceinline__ v16bf load_a_frag_f32(const float* __restrict__ x, int stride,
                                                 int mrow, int k0, int hs) {
    const float* base = x + (size_t)mrow * stride + k0 + hs * 8;
    v16bf a;
#pragma unroll
    for (int j = 0; j < 8; ++j) a[j] = f2bf(base[j]);
#pragma unroll
    for (int j = 0; j < 8; ++j) a[j + 8] = f2bf(base[j + 16]);
    return a;
}

// A fragment from bf16 row-major tile: two contiguous 16B vector loads.
// Works for both global and LDS tiles (addrspace inferred).
__device__ __forceinline__ v16bf load_a_frag_bf(const __bf16* x, int stride,
                                                int mrow, int k0, int hs) {
    const __bf16* base = x + (size_t)mrow * stride + k0 + hs * 8;
    v8bf lo = *(const v8bf*)(base);
    v8bf hi = *(const v8bf*)(base + 16);
    v16bf a;
#pragma unroll
    for (int j = 0; j < 8; ++j) { a[j] = lo[j]; a[j + 8] = hi[j]; }
    return a;
}

// B fragment (32x16 bf16) from bf16 weights stored [N,K] row-major.
__device__ __forceinline__ v16bf load_b_frag(const __bf16* __restrict__ W, int K,
                                             int ncol, int k0, int hs) {
    const __bf16* base = W + (size_t)ncol * K + k0 + hs * 8;
    v8bf lo = *(const v8bf*)(base);
    v8bf hi = *(const v8bf*)(base + 16);
    v16bf b;
#pragma unroll
    for (int j = 0; j < 8; ++j) { b[j] = lo[j]; b[j + 8] = hi[j]; }
    return b;
}

// ---------------------------------------------------------------------------
// Tensor Data Mover staging: contiguous 1-D bf16 tile global -> LDS.
// D# per CDNA5 ISA: group0 {count=1, lds_addr, global_addr,
// type=2}; group1 {data_size=2B, tensor_dim0=nelem, tensor_dim1=1,
// tile_dim0=nelem, tile_dim1=1, dim0_stride=nelem}.  Issued by wave 0 only
// (TDM ignores EXEC; TENSORcnt tracks completion). Fallback: vector copy.
// ---------------------------------------------------------------------------
__device__ __forceinline__ void tdm_stage(const __bf16* gsrc, __bf16* sdst, unsigned nelem) {
#if __has_builtin(__builtin_amdgcn_tensor_load_to_lds)
    if ((threadIdx.x >> 5) == 0) {
        unsigned long long ga = (unsigned long long)(uintptr_t)gsrc;
        unsigned lds_off = (unsigned)(uintptr_t)sdst;   // LDS aperture: low 32 bits = byte offset
        u32x4 g0; i32x8 g1; i32x4 gz4 = {0, 0, 0, 0};
        g0[0] = 1u;                                     // count=1 (valid user D#)
        g0[1] = lds_off;                                // lds_addr (bytes)
        g0[2] = (unsigned)ga;                           // global_addr[31:0]
        g0[3] = (unsigned)((ga >> 32) & 0x01FFFFFFull) | 0x80000000u;  // addr[56:32] | type=2
        g1[0] = 0x10000;                                // data_size=1 -> 2 bytes
        g1[1] = (int)((nelem & 0xFFFFu) << 16);         // tensor_dim0[15:0] @ bit48
        g1[2] = (int)((nelem >> 16) & 0xFFFFu) | (1 << 16); // tensor_dim0[31:16] | tensor_dim1=1
        g1[3] = (int)(nelem << 16);                     // tile_dim0 @ bits 127:112
        g1[4] = 1;                                      // tile_dim1 = 1
        g1[5] = (int)nelem;                             // tensor_dim0_stride[31:0]
        g1[6] = 0; g1[7] = 0;
#if defined(__clang_major__) && __clang_major__ >= 23
        i32x8 gz8 = {0, 0, 0, 0, 0, 0, 0, 0};
        __builtin_amdgcn_tensor_load_to_lds(g0, g1, gz4, gz4, gz8, 0);
#else
        __builtin_amdgcn_tensor_load_to_lds(g0, g1, gz4, gz4, 0);
#endif
    }
#else
    for (unsigned i = threadIdx.x * 8u; i < nelem; i += 256u * 8u)
        *(v8bf*)(sdst + i) = *(const v8bf*)(gsrc + i);
#endif
}

__device__ __forceinline__ void tdm_wait() {
#if __has_builtin(__builtin_amdgcn_s_wait_tensorcnt)
    if ((threadIdx.x >> 5) == 0) __builtin_amdgcn_s_wait_tensorcnt(0);
#endif
    __syncthreads();
}

// ---------------------------------------------------------------------------
// Utility kernels
// ---------------------------------------------------------------------------
__global__ void f32_to_bf16_kernel(const float* __restrict__ in, __bf16* __restrict__ out, int n) {
    int i = blockIdx.x * blockDim.x + threadIdx.x;
    if (i < n) out[i] = f2bf(in[i]);
}

__global__ void zero_f32_kernel(float* __restrict__ p, int n) {
    int i = blockIdx.x * blockDim.x + threadIdx.x;
    if (i < n) p[i] = 0.0f;
}

__global__ void zero_u16_kernel(unsigned short* __restrict__ p, int n) {
    int i = blockIdx.x * blockDim.x + threadIdx.x;
    if (i < n) p[i] = 0;
}

__global__ void lengths_kernel(const int* __restrict__ umask, int* __restrict__ L) {
    int b = threadIdx.x;
    if (b >= 64) return;
    int s = 0;
    for (int t = 0; t < 200; ++t) s += umask[b * 200 + t];
    L[b] = s;
}

// out[t,b,:] = (L[b]-1-t >= 0) ? in[L[b]-1-t, b, :] : 0        (shape [200,64,D])
__global__ void reverse_seq_kernel(const float* __restrict__ in, float* __restrict__ out,
                                   const int* __restrict__ L, int D) {
    size_t i = (size_t)blockIdx.x * blockDim.x + threadIdx.x;
    size_t total = (size_t)200 * 64 * D;
    if (i >= total) return;
    int d = (int)(i % D);
    size_t r = i / D;
    int b = (int)(r % 64);
    int t = (int)(r / 64);
    int idx = L[b] - 1 - t;
    out[i] = (idx >= 0) ? in[((size_t)idx * 64 + b) * D + d] : 0.0f;
}

// ---------------------------------------------------------------------------
// Generic WMMA GEMM (fp32 A with in-register bf16 conversion):
//   out[M,N] = concat_K(a0..a3)[M,K] @ W[N,K]^T + bias ; optional relu+residual
// ---------------------------------------------------------------------------
__global__ void __launch_bounds__(256)
wmma_gemm_kernel(const float* a0, const float* a1, const float* a2, const float* a3,
                 int Kw, int K, const __bf16* __restrict__ W,
                 const float* __restrict__ bias, const float* __restrict__ residual,
                 float* __restrict__ out, int N, int do_relu) {
    const float* srcs[4] = {a0, a1, a2, a3};
    int lane = threadIdx.x & 31;
    int wid  = blockIdx.x * (blockDim.x >> 5) + (threadIdx.x >> 5);
    int ntiles = N >> 4;
    int tile_m = wid / ntiles;
    int tile_n = wid % ntiles;
    int hs   = lane >> 4;
    int nloc = lane & 15;
    int mrow = tile_m * 16 + nloc;
    int ncol = tile_n * 16 + nloc;
    v8f acc = {};
    for (int k0 = 0; k0 < K; k0 += 32) {
        int s  = k0 / Kw;
        int kk = k0 - s * Kw;
        v16bf a = load_a_frag_f32(srcs[s], Kw, mrow, kk, hs);
        v16bf b = load_b_frag(W, K, ncol, k0, hs);
        acc = wmma_bf16(a, b, acc);
    }
    float bv = bias[ncol];
    int row0 = tile_m * 16 + hs * 8;
    if (residual) {
#pragma unroll
        for (int r = 0; r < 8; ++r) {
            float v = acc[r] + bv;
            if (do_relu) v = fmaxf(v, 0.0f);
            v += residual[(size_t)(row0 + r) * N + ncol];
            out[(size_t)(row0 + r) * N + ncol] = v;
        }
    } else {
#pragma unroll
        for (int r = 0; r < 8; ++r) {
            float v = acc[r] + bv;
            if (do_relu) v = fmaxf(v, 0.0f);
            out[(size_t)(row0 + r) * N + ncol] = v;
        }
    }
}

// ---------------------------------------------------------------------------
// Fused GRU step (bf16 activations): gi = x@Wih^T + bih ; gh = h@Whh^T + bhh.
// Hidden dim 256 (768 gate columns). Block = 8 waves, all sharing one A row
// tile (tile_m = blockIdx>>1): x tile [16,Kx] and h tile [16,256] are staged
// into LDS once via the Tensor Data Mover, then each wave computes one 16x16
// tile of h_new with 6 WMMA accumulators. Weights stream from L2.
// ---------------------------------------------------------------------------
__global__ void __launch_bounds__(256)
gru_fused_kernel(const __bf16* __restrict__ x, int Kx,
                 const __bf16* __restrict__ hbf,     // bf16 copy of prev hidden [M,256]
                 const float*  __restrict__ hfp,     // fp32 prev hidden (for z*h term)
                 const __bf16* __restrict__ Wih, const __bf16* __restrict__ Whh,
                 const float* __restrict__ bih, const float* __restrict__ bhh,
                 float* __restrict__ h_out, float* __restrict__ h_out2,
                 __bf16* __restrict__ h_out_bf,
                 float* __restrict__ score_out, const float* __restrict__ attn_w) {
    __shared__ __attribute__((aligned(16))) __bf16 sx[16 * 1024];
    __shared__ __attribute__((aligned(16))) __bf16 sh[16 * 256];

    int lane = threadIdx.x & 31;
    int wave = threadIdx.x >> 5;
    int tile_m = blockIdx.x >> 1;            // one A-row tile per block pair
    int tile_n = (blockIdx.x & 1) * 8 + wave;
    int hs   = lane >> 4;
    int nloc = lane & 15;
    int lrow = nloc;                         // row within the 16-row LDS tile
    int nc   = tile_n * 16 + nloc;           // column within hidden dim [0,256)

    // Stage A tiles (contiguous 16 rows) into LDS via TDM, then wait TENSORcnt.
    tdm_stage(x   + (size_t)tile_m * 16 * Kx,  sx, (unsigned)(16 * Kx));
    tdm_stage(hbf + (size_t)tile_m * 16 * 256, sh, 16 * 256);
    tdm_wait();

    v8f ai0 = {}, ai1 = {}, ai2 = {}, ah0 = {}, ah1 = {}, ah2 = {};

    for (int k0 = 0; k0 < Kx; k0 += 32) {
        v16bf a  = load_a_frag_bf(sx, Kx, lrow, k0, hs);
        if (k0 + 32 < Kx)
            __builtin_prefetch(Wih + (size_t)nc * Kx + k0 + 32, 0, 1);
        v16bf b0 = load_b_frag(Wih, Kx, nc,        k0, hs);
        v16bf b1 = load_b_frag(Wih, Kx, nc + 256,  k0, hs);
        v16bf b2 = load_b_frag(Wih, Kx, nc + 512,  k0, hs);
        ai0 = wmma_bf16(a, b0, ai0);
        ai1 = wmma_bf16(a, b1, ai1);
        ai2 = wmma_bf16(a, b2, ai2);
    }
#pragma unroll 2
    for (int k0 = 0; k0 < 256; k0 += 32) {
        v16bf a  = load_a_frag_bf(sh, 256, lrow, k0, hs);
        v16bf b0 = load_b_frag(Whh, 256, nc,        k0, hs);
        v16bf b1 = load_b_frag(Whh, 256, nc + 256,  k0, hs);
        v16bf b2 = load_b_frag(Whh, 256, nc + 512,  k0, hs);
        ah0 = wmma_bf16(a, b0, ah0);
        ah1 = wmma_bf16(a, b1, ah1);
        ah2 = wmma_bf16(a, b2, ah2);
    }

    float bi0 = bih[nc], bi1 = bih[256 + nc], bi2 = bih[512 + nc];
    float bh0 = bhh[nc], bh1 = bhh[256 + nc], bh2 = bhh[512 + nc];
    float aw  = score_out ? attn_w[nc] : 0.0f;

#pragma unroll
    for (int r = 0; r < 8; ++r) {
        int row = tile_m * 16 + hs * 8 + r;
        float gir = ai0[r] + bi0, giz = ai1[r] + bi1, gin = ai2[r] + bi2;
        float ghr = ah0[r] + bh0, ghz = ah1[r] + bh1, ghn = ah2[r] + bh2;
        float rg = 1.0f / (1.0f + expf(-(gir + ghr)));
        float zg = 1.0f / (1.0f + expf(-(giz + ghz)));
        float ng = tanhf(gin + rg * ghn);
        float hp = hfp[(size_t)row * 256 + nc];
        float hn = (1.0f - zg) * ng + zg * hp;
        h_out[(size_t)row * 256 + nc] = hn;
        if (h_out2)    h_out2[(size_t)row * 256 + nc] = hn;
        if (h_out_bf)  h_out_bf[(size_t)row * 256 + nc] = f2bf(hn);
        if (score_out) atomicAdd(score_out + row, hn * aw);
    }
}

// ---------------------------------------------------------------------------
// Per-step helper kernels (emit bf16 GEMM inputs directly)
// ---------------------------------------------------------------------------
__global__ void prep_g_kernel(const float* __restrict__ qmask_dir,
                              const float* __restrict__ q0, const float* __restrict__ r0,
                              const float* __restrict__ u_seq,
                              __bf16* __restrict__ xg, int* __restrict__ qi_buf,
                              float* __restrict__ score_slot, int t) {
    int i = blockIdx.x * blockDim.x + threadIdx.x;
    if (i >= 64 * 1024) return;
    int b = i >> 10, k = i & 1023;
    const float* qm = qmask_dir + (size_t)t * 128 + b * 2;
    int qi = (qm[1] > qm[0]) ? 1 : 0;
    float v;
    if (k < 256)      v = q0[(b * 2 + qi) * 256 + k];
    else if (k < 512) v = r0[(b * 2 + qi) * 256 + (k - 256)];
    else              v = u_seq[((size_t)t * 64 + b) * 512 + (k - 512)];
    xg[i] = f2bf(v);
    if (k == 0) { qi_buf[b] = qi; score_slot[b] = 0.0f; }
}

__global__ void attn_kernel(const float* __restrict__ score_buf,
                            const float* __restrict__ g_buf,
                            float* __restrict__ c_, int t) {
    int b = blockIdx.x, tid = threadIdx.x;
    __shared__ float red[256];
    if (t == 0) { c_[b * 256 + tid] = 0.0f; return; }
    float m = -1e30f;
    for (int tt = tid; tt < t; tt += 256) m = fmaxf(m, score_buf[tt * 64 + b]);
    red[tid] = m; __syncthreads();
    for (int s = 128; s > 0; s >>= 1) { if (tid < s) red[tid] = fmaxf(red[tid], red[tid + s]); __syncthreads(); }
    m = red[0]; __syncthreads();
    float ss = 0.0f;
    for (int tt = tid; tt < t; tt += 256) ss += expf(score_buf[tt * 64 + b] - m);
    red[tid] = ss; __syncthreads();
    for (int s = 128; s > 0; s >>= 1) { if (tid < s) red[tid] += red[tid + s]; __syncthreads(); }
    float inv = 1.0f / red[0];
    int d = tid;
    float acc = 0.0f;
    for (int tt = 0; tt < t; ++tt) {
        float al = expf(score_buf[tt * 64 + b] - m) * inv;
        acc += al * g_buf[(size_t)tt * 64 * 256 + b * 256 + d];
    }
    c_[b * 256 + d] = acc;
}

__global__ void prep_rp_kernel(const float* __restrict__ c_, const float* __restrict__ g_new,
                               const float* __restrict__ u_seq,
                               __bf16* __restrict__ inp_r, __bf16* __restrict__ inp_p, int t) {
    int i = blockIdx.x * blockDim.x + threadIdx.x;
    if (i >= 128 * 768) return;
    int m = i / 768, k = i % 768, b = m >> 1;
    __bf16 uv = (k >= 256) ? f2bf(u_seq[((size_t)t * 64 + b) * 512 + (k - 256)]) : f2bf(0.0f);
    inp_r[i] = (k < 256) ? f2bf(c_[b * 256 + k])    : uv;
    inp_p[i] = (k < 256) ? f2bf(g_new[b * 256 + k]) : uv;
}

__global__ void combine_kernel(const float* __restrict__ qmask_dir,
                               const float* __restrict__ ql_, const float* __restrict__ qs_,
                               const float* __restrict__ rs_,
                               float* __restrict__ q0, float* __restrict__ r0,
                               __bf16* __restrict__ q0bf, __bf16* __restrict__ r0bf, int t) {
    int i = blockIdx.x * blockDim.x + threadIdx.x;
    if (i >= 128 * 256) return;
    int m = i >> 8;
    float qm = qmask_dir[(size_t)t * 128 + m];
    float qv = ql_[i] * qm + qs_[i] * (1.0f - qm);
    float rv = rs_[i] * qm;
    q0[i] = qv;  r0[i] = rv;
    q0bf[i] = f2bf(qv);  r0bf[i] = f2bf(rv);
}

__global__ void prep_e_kernel(const float* __restrict__ u_seq, const float* __restrict__ q0,
                              const float* __restrict__ g_new, const int* __restrict__ qi_buf,
                              __bf16* __restrict__ xe, int t) {
    int i = blockIdx.x * blockDim.x + threadIdx.x;
    if (i >= 64 * 1024) return;
    int b = i >> 10, k = i & 1023;
    float v;
    if (k < 512)      v = u_seq[((size_t)t * 64 + b) * 512 + k];
    else if (k < 768) v = q0[(b * 2 + qi_buf[b]) * 256 + (k - 512)];
    else              v = g_new[b * 256 + (k - 768)];
    xe[i] = f2bf(v);
}

__global__ void __launch_bounds__(256)
cls_kernel(const float* __restrict__ hidden, const float* __restrict__ Ws,
           const float* __restrict__ bs, float* __restrict__ out) {
    int row  = blockIdx.x * 8 + (threadIdx.x >> 5);
    int lane = threadIdx.x & 31;
    if (row >= 12800) return;
    float logit[7];
#pragma unroll
    for (int c = 0; c < 7; ++c) {
        float p = 0.0f;
        for (int k = lane; k < 512; k += 32)
            p += hidden[(size_t)row * 512 + k] * Ws[c * 512 + k];
        for (int off = 16; off > 0; off >>= 1) p += __shfl_xor(p, off, 32);
        logit[c] = p + bs[c];
    }
    if (lane == 0) {
        float m = logit[0];
#pragma unroll
        for (int c = 1; c < 7; ++c) m = fmaxf(m, logit[c]);
        float s = 0.0f;
#pragma unroll
        for (int c = 0; c < 7; ++c) s += expf(logit[c] - m);
        float lse = m + logf(s);
#pragma unroll
        for (int c = 0; c < 7; ++c) out[(size_t)row * 7 + c] = logit[c] - lse;
    }
}

// ---------------------------------------------------------------------------
// Host orchestration
// ---------------------------------------------------------------------------
// Input index map (setup_inputs() dict insertion order, nested dicts in order):
// 0:r1 1:r2 2:r3 3:r4 4:qmask 5:umask 6:Win 7:bin
// f cell @8:  g{Wih,Whh,bih,bhh}=8..11  p=12..15  pl=16..19  r=20..23  e=24..27  attn_w=28
// b cell @29: g=29..32 p=33..36 pl=37..40 r=41..44 e=45..48 attn_w=49
// 50:Wl 51:bl 52:Ws 53:bs
extern "C" void kernel_launch(void* const* d_in, const int* in_sizes, int n_in,
                              void* d_out, int out_size, void* d_ws, size_t ws_size,
                              hipStream_t stream) {
    (void)in_sizes; (void)n_in; (void)out_size; (void)ws_size;
    const int T = 200;
    const size_t TB = (size_t)T * 64;            // 12800

    char* wsb = (char*)d_ws;
    size_t off = 0;
    auto alloc = [&](size_t bytes) -> void* {
        void* p = wsb + off;
        off = (off + bytes + 255) & ~(size_t)255;
        return p;
    };

    // --- bf16 weight conversion ---
    struct WJob { int idx; int elems; };
    const WJob jobs[] = {
        {6, 512 * 2048},
        {8, 768 * 1024}, {9, 768 * 256}, {12, 768 * 768}, {13, 768 * 256},
        {16, 768 * 768}, {17, 768 * 256}, {20, 768 * 768}, {21, 768 * 256},
        {24, 768 * 1024}, {25, 768 * 256},
        {29, 768 * 1024}, {30, 768 * 256}, {33, 768 * 768}, {34, 768 * 256},
        {37, 768 * 768}, {38, 768 * 256}, {41, 768 * 768}, {42, 768 * 256},
        {45, 768 * 1024}, {46, 768 * 256},
        {50, 512 * 512},
    };
    __bf16* wbf[54] = {};
    for (const WJob& j : jobs) {
        __bf16* p = (__bf16*)alloc((size_t)j.elems * 2);
        f32_to_bf16_kernel<<<(j.elems + 255) / 256, 256, 0, stream>>>(
            (const float*)d_in[j.idx], p, j.elems);
        wbf[j.idx] = p;
    }

    // --- workspace buffers ---
    float*  r_seq    = (float*)alloc(TB * 512 * 4);
    float*  rev_r    = (float*)alloc(TB * 512 * 4);
    float*  rev_q    = (float*)alloc(TB * 2 * 4);
    int*    L        = (int*)  alloc(64 * 4);
    float*  es_f     = (float*)alloc(TB * 256 * 4);
    float*  es_b_raw = (float*)alloc(TB * 256 * 4);
    float*  es_b     = (float*)alloc(TB * 256 * 4);
    float*  hidden   = (float*)alloc(TB * 512 * 4);
    float*  g_buf    = (float*)alloc((size_t)T * 64 * 256 * 4);
    float*  score    = (float*)alloc((size_t)T * 64 * 4);
    float*  gA       = (float*)alloc(64 * 256 * 4);
    float*  gB       = (float*)alloc(64 * 256 * 4);
    float*  eA       = (float*)alloc(64 * 256 * 4);
    float*  eB       = (float*)alloc(64 * 256 * 4);
    __bf16* gbfA     = (__bf16*)alloc(64 * 256 * 2);
    __bf16* gbfB     = (__bf16*)alloc(64 * 256 * 2);
    __bf16* ebfA     = (__bf16*)alloc(64 * 256 * 2);
    __bf16* ebfB     = (__bf16*)alloc(64 * 256 * 2);
    float*  q0       = (float*)alloc(128 * 256 * 4);
    float*  r0       = (float*)alloc(128 * 256 * 4);
    __bf16* q0bf     = (__bf16*)alloc(128 * 256 * 2);
    __bf16* r0bf     = (__bf16*)alloc(128 * 256 * 2);
    float*  c_       = (float*)alloc(64 * 256 * 4);
    __bf16* xg       = (__bf16*)alloc(64 * 1024 * 2);
    __bf16* xe       = (__bf16*)alloc(64 * 1024 * 2);
    __bf16* inp_r    = (__bf16*)alloc(128 * 768 * 2);
    __bf16* inp_p    = (__bf16*)alloc(128 * 768 * 2);
    float*  rs_      = (float*)alloc(128 * 256 * 4);
    float*  qs_      = (float*)alloc(128 * 256 * 4);
    float*  ql_      = (float*)alloc(128 * 256 * 4);
    int*    qi_buf   = (int*)  alloc(64 * 4);

    auto F = [&](int i) { return (const float*)d_in[i]; };
    auto zerof = [&](float* p, int n) {
        zero_f32_kernel<<<(n + 255) / 256, 256, 0, stream>>>(p, n);
    };
    auto zerob = [&](__bf16* p, int n) {
        zero_u16_kernel<<<(n + 255) / 256, 256, 0, stream>>>((unsigned short*)p, n);
    };
    auto gru = [&](const __bf16* x, int Kx, const __bf16* hbf, const float* hfp,
                   const __bf16* Wih, const __bf16* Whh,
                   const float* bih, const float* bhh,
                   float* h_out, float* h_out2, __bf16* h_out_bf,
                   float* score_out, const float* attn_w, int M) {
        int blocks = (M / 16) * 2;   // block pair per A-row tile, 8 waves/block
        gru_fused_kernel<<<blocks, 256, 0, stream>>>(x, Kx, hbf, hfp, Wih, Whh, bih, bhh,
                                                     h_out, h_out2, h_out_bf,
                                                     score_out, attn_w);
    };

    // --- input projection: r = concat(r1..r4) @ Win^T + bin ---
    wmma_gemm_kernel<<<(int)(((TB / 16) * (512 / 16)) / 8), 256, 0, stream>>>(
        F(0), F(1), F(2), F(3), 512, 2048, wbf[6], F(7), nullptr, r_seq, 512, 0);

    // --- lengths, reversed sequences for backward direction ---
    lengths_kernel<<<1, 64, 0, stream>>>((const int*)d_in[5], L);
    reverse_seq_kernel<<<(int)((TB * 512 + 255) / 256), 256, 0, stream>>>(r_seq, rev_r, L, 512);
    reverse_seq_kernel<<<(int)((TB * 2 + 255) / 256), 256, 0, stream>>>(F(4), rev_q, L, 2);

    // --- bidirectional emotion RNN ---
    for (int dir = 0; dir < 2; ++dir) {
        const int cb = (dir == 0) ? 8 : 29;
        const float* u_seq     = (dir == 0) ? r_seq : rev_r;
        const float* qmask_dir = (dir == 0) ? F(4) : rev_q;
        float*       es_out    = (dir == 0) ? es_f : es_b_raw;
        const float* attn_w    = F(cb + 20);

        zerof(gA, 64 * 256); zerof(q0, 128 * 256); zerof(r0, 128 * 256); zerof(eA, 64 * 256);
        zerob(gbfA, 64 * 256); zerob(ebfA, 64 * 256);
        zerob(q0bf, 128 * 256); zerob(r0bf, 128 * 256);

        float  *gIn = gA, *gOut = gB, *eIn = eA, *eOut = eB;
        __bf16 *gbIn = gbfA, *gbOut = gbfB, *ebIn = ebfA, *ebOut = ebfB;

        for (int t = 0; t < T; ++t) {
            prep_g_kernel<<<256, 256, 0, stream>>>(qmask_dir, q0, r0, u_seq,
                                                   xg, qi_buf, score + t * 64, t);
            gru(xg, 1024, gbIn, gIn, wbf[cb + 0], wbf[cb + 1], F(cb + 2), F(cb + 3),
                gOut, g_buf + (size_t)t * 64 * 256, gbOut, score + t * 64, attn_w, 64);
            attn_kernel<<<64, 256, 0, stream>>>(score, g_buf, c_, t);
            prep_rp_kernel<<<384, 256, 0, stream>>>(c_, gOut, u_seq, inp_r, inp_p, t);
            gru(inp_r, 768, r0bf, r0, wbf[cb + 12], wbf[cb + 13], F(cb + 14), F(cb + 15),
                rs_, nullptr, nullptr, nullptr, nullptr, 128);
            gru(inp_p, 768, q0bf, q0, wbf[cb + 4], wbf[cb + 5], F(cb + 6), F(cb + 7),
                qs_, nullptr, nullptr, nullptr, nullptr, 128);
            gru(inp_p, 768, q0bf, q0, wbf[cb + 8], wbf[cb + 9], F(cb + 10), F(cb + 11),
                ql_, nullptr, nullptr, nullptr, nullptr, 128);
            combine_kernel<<<128, 256, 0, stream>>>(qmask_dir, ql_, qs_, rs_,
                                                    q0, r0, q0bf, r0bf, t);
            prep_e_kernel<<<256, 256, 0, stream>>>(u_seq, q0, gOut, qi_buf, xe, t);
            gru(xe, 1024, ebIn, eIn, wbf[cb + 16], wbf[cb + 17], F(cb + 18), F(cb + 19),
                eOut, es_out + (size_t)t * 64 * 256, ebOut, nullptr, nullptr, 64);
            { float*  tmp = gIn;  gIn  = gOut;  gOut  = tmp; }
            { float*  tmp = eIn;  eIn  = eOut;  eOut  = tmp; }
            { __bf16* tmp = gbIn; gbIn = gbOut; gbOut = tmp; }
            { __bf16* tmp = ebIn; ebIn = ebOut; ebOut = tmp; }
        }
    }

    // --- un-reverse backward outputs ---
    reverse_seq_kernel<<<(int)((TB * 256 + 255) / 256), 256, 0, stream>>>(es_b_raw, es_b, L, 256);

    // --- hidden = relu(concat(ef,eb) @ Wl^T + bl) + r ---
    wmma_gemm_kernel<<<(int)(((TB / 16) * (512 / 16)) / 8), 256, 0, stream>>>(
        es_f, es_b, es_b, es_b, 256, 512, wbf[50], F(51), r_seq, hidden, 512, 1);

    // --- logits + log_softmax ---
    cls_kernel<<<(int)(TB / 8), 256, 0, stream>>>(hidden, F(52), F(53), (float*)d_out);
}
